// COAMultiHeadAttention_30485677867269
// MI455X (gfx1250) — compile-verified, matrix operand order checked by reading, not run on hardware
//
#include <hip/hip_runtime.h>

typedef __attribute__((ext_vector_type(16))) __bf16 v16bf;
typedef __attribute__((ext_vector_type(8)))  float  v8f;
typedef unsigned short u16;
typedef unsigned int   u32;

#define D_MODEL  1024
#define N_HEADS  16
#define HEAD_DIM 64
#define B_SZ     2
#define T_SEQ    2048
#define M_TOK    (B_SZ * T_SEQ)     // 4096 token rows
#define TW       (T_SEQ / 32)       // 64 mask words per row

// ---------- helpers ----------

static __device__ __forceinline__ u16 f32_to_bf16(float f) {
  u32 x = __builtin_bit_cast(u32, f);
  u32 r = x + 0x7fffu + ((x >> 16) & 1u);   // round-to-nearest-even
  return (u16)(r >> 16);
}

static __device__ __forceinline__ u32 pack_bf16x2(float lo, float hi) {
  return (u32)f32_to_bf16(lo) | ((u32)f32_to_bf16(hi) << 16);
}

union FragU { v16bf v; uint4 q[2]; u16 h[16]; };

// 16-bit A/B fragment layout (ISA 7.12.2): per lane, elements 0..7 are K=+0..7,
// elements 8..15 are K=+16..23 (caller pre-adds the hi-lane +8 K offset).
static __device__ __forceinline__ v16bf load_frag_bf16(const u16* __restrict__ p) {
  FragU f;
  f.q[0] = *(const uint4*)(p + 0);
  f.q[1] = *(const uint4*)(p + 16);
  return f.v;
}

// ---------- fp32 -> bf16 bulk conversion (bandwidth-bound) ----------
__global__ __launch_bounds__(256)
void cvt_bf16_k(const float* __restrict__ src, u16* __restrict__ dst, int n8)
{
  const int i = blockIdx.x * blockDim.x + threadIdx.x;
  if (i >= n8) return;
  float4 a = ((const float4*)src)[2 * i + 0];
  float4 b = ((const float4*)src)[2 * i + 1];
  uint4 o;
  o.x = pack_bf16x2(a.x, a.y);
  o.y = pack_bf16x2(a.z, a.w);
  o.z = pack_bf16x2(b.x, b.y);
  o.w = pack_bf16x2(b.z, b.w);
  ((uint4*)dst)[i] = o;
}

// ---------- int32 mask -> bitmask (1 bit per entry) via wave ballot ----------
__global__ __launch_bounds__(256)
void mask_pack_k(const int* __restrict__ mask, u32* __restrict__ bits, int nwords)
{
  const int lane = threadIdx.x & 31;
  const int gw   = (blockIdx.x * blockDim.x + threadIdx.x) >> 5;   // wave id = word id
  if (gw >= nwords) return;
  const int v = mask[(size_t)gw * 32 + lane];
  const unsigned long long bal = __ballot(v != 0);
  if (lane == 0) bits[gw] = (u32)bal;
}

// ---------- GEMM: C[M,N] = A[M,K] * W[N,K]^T + bias  (A, W already bf16) ----------
// OUT_MODE: 0 = bf16 out, layout [B,H,T,HD]   (Q, K)
//           1 = bf16 out, layout [B,H,HD,T]   (V transposed)
//           2 = f32  out, layout [M, D_MODEL] (final projection)
template <int OUT_MODE>
__global__ __launch_bounds__(256)
void gemm_wmma_k(const u16* __restrict__ A, const u16* __restrict__ W,
                 const float* __restrict__ bias, void* __restrict__ Out)
{
  const int K    = D_MODEL;
  const int lane = threadIdx.x & 31;
  const int wave = threadIdx.x >> 5;
  const int c16  = lane & 15;
  const int hi   = lane >> 4;
  const int koff = hi ? 8 : 0;

  const int row0 = blockIdx.y * 128 + (wave >> 1) * 32;  // 4 waves down
  const int col0 = blockIdx.x * 128 + (wave & 1) * 64;   // 2 waves across

  const v8f vzero = {};
  v8f acc[2][4];
#pragma unroll
  for (int i = 0; i < 2; ++i)
#pragma unroll
    for (int j = 0; j < 4; ++j) acc[i][j] = vzero;

  // hot loop: 12x global_load_b128 + 8x v_wmma per 32-deep K step
  for (int k0 = 0; k0 < K; k0 += 32) {
    v16bf a[2], b[4];
#pragma unroll
    for (int i = 0; i < 2; ++i) {
      const int r = row0 + 16 * i + c16;
      a[i] = load_frag_bf16(A + r * K + k0 + koff);
    }
#pragma unroll
    for (int j = 0; j < 4; ++j) {
      const int n = col0 + 16 * j + c16;
      b[j] = load_frag_bf16(W + n * K + k0 + koff);   // torch weight: row n, contiguous K
    }
#pragma unroll
    for (int i = 0; i < 2; ++i)
#pragma unroll
      for (int j = 0; j < 4; ++j)
        acc[i][j] = __builtin_amdgcn_wmma_f32_16x16x32_bf16(
            false, a[i], false, b[j], (short)0, acc[i][j], false, false);
  }

  // epilogue: C layout -> lane holds column n, rows (hi*8 + r)
#pragma unroll
  for (int i = 0; i < 2; ++i) {
#pragma unroll
    for (int j = 0; j < 4; ++j) {
      const int n = col0 + 16 * j + c16;
      const float bv = bias[n];
#pragma unroll
      for (int r = 0; r < 8; ++r) {
        const int m = row0 + 16 * i + hi * 8 + r;
        const float v = acc[i][j][r] + bv;
        if constexpr (OUT_MODE == 2) {
          ((float*)Out)[(size_t)m * D_MODEL + n] = v;
        } else {
          const int bb = m >> 11;              // / T_SEQ
          const int t  = m & (T_SEQ - 1);
          const int h  = n >> 6;               // / HEAD_DIM
          const int d  = n & (HEAD_DIM - 1);
          const u16 hv = f32_to_bf16(v);
          if constexpr (OUT_MODE == 0)
            ((u16*)Out)[(size_t)(((bb * N_HEADS + h) * T_SEQ) + t) * HEAD_DIM + d] = hv;
          else
            ((u16*)Out)[(size_t)(((bb * N_HEADS + h) * HEAD_DIM) + d) * T_SEQ + t] = hv;
        }
      }
    }
  }
}

// ---------- Flash attention: one wave per (b, h, 16 query rows) ----------
__global__ __launch_bounds__(256)
void flash_attn_k(const u16* __restrict__ Qh, const u16* __restrict__ Kh,
                  const u16* __restrict__ Vt, const u32* __restrict__ mbits,
                  u16* __restrict__ att)
{
  __shared__ u16 lds[8][16 * 32];   // per-wave 16x32 bf16 P tile (1 KB each)

  const int lane = threadIdx.x & 31;
  const int wave = threadIdx.x >> 5;
  const int c16  = lane & 15;
  const int hi   = lane >> 4;
  const int koff = hi ? 8 : 0;

  const int gw = blockIdx.x * 8 + wave;          // 0 .. 4095
  const int qt = gw & (T_SEQ / 16 - 1);          // 128 q-tiles
  const int h  = (gw >> 7) & (N_HEADS - 1);
  const int b  = gw >> 11;
  const int q0 = qt * 16;

  const u16* Qb = Qh + (size_t)((b * N_HEADS + h) * T_SEQ) * HEAD_DIM;
  const u16* Kb = Kh + (size_t)((b * N_HEADS + h) * T_SEQ) * HEAD_DIM;
  const u16* Vb = Vt + (size_t)((b * N_HEADS + h) * HEAD_DIM) * T_SEQ;
  // this lane tracks the mask bit-words of query row (q0 + lane&15)
  const u32* MbW = mbits + (size_t)(b * T_SEQ + q0 + c16) * TW;
  const u32 bit0 = 1u << c16;          // tile cols 0..15
  const u32 bit1 = 0x10000u << c16;    // tile cols 16..31

  // Q fragments (16 q rows x 64 d), held for whole loop
  v16bf qf[2];
#pragma unroll
  for (int dk = 0; dk < 2; ++dk)
    qf[dk] = load_frag_bf16(Qb + (q0 + c16) * HEAD_DIM + dk * 32 + koff);

  const v8f vzero = {};
  v8f o[4];
#pragma unroll
  for (int t = 0; t < 4; ++t) o[t] = vzero;
  float mrow[8], lrow[8];
#pragma unroll
  for (int r = 0; r < 8; ++r) { mrow[r] = -1e30f; lrow[r] = 0.0f; }

  u16* P = &lds[wave][0];

  for (int kt = 0; kt < T_SEQ; kt += 32) {
    // S = Q * K^T for 32 key columns (two 16x16 C tiles)
    v8f s0 = vzero, s1 = vzero;
#pragma unroll
    for (int dk = 0; dk < 2; ++dk) {
      v16bf kb0 = load_frag_bf16(Kb + (kt + c16) * HEAD_DIM + dk * 32 + koff);
      v16bf kb1 = load_frag_bf16(Kb + (kt + 16 + c16) * HEAD_DIM + dk * 32 + koff);
      s0 = __builtin_amdgcn_wmma_f32_16x16x32_bf16(false, qf[dk], false, kb0, (short)0, s0, false, false);
      s1 = __builtin_amdgcn_wmma_f32_16x16x32_bf16(false, qf[dk], false, kb1, (short)0, s1, false, false);
    }

    // one mask bit-word per lane covers this 32-column tile for its row
    const u32 wrow = MbW[kt >> 5];

    const float scale = 0.125f;    // 1/sqrt(HEAD_DIM)
    float p0[8], p1[8], tmax[8];
#pragma unroll
    for (int r = 0; r < 8; ++r) {
      const u32 wr = (u32)__shfl((int)wrow, hi * 8 + r, 32);   // word of row (hi*8+r)
      float v0 = s0[r] * scale;
      float v1 = s1[r] * scale;
      if ((wr & bit0) == 0) v0 = -1e30f;
      if ((wr & bit1) == 0) v1 = -1e30f;
      p0[r] = v0; p1[r] = v1;
      tmax[r] = fmaxf(v0, v1);
    }
    // row max across the 16 lanes holding this row's columns
#pragma unroll
    for (int xm = 1; xm < 16; xm <<= 1)
#pragma unroll
      for (int r = 0; r < 8; ++r)
        tmax[r] = fmaxf(tmax[r], __shfl_xor(tmax[r], xm, 32));

    float alpha[8];
#pragma unroll
    for (int r = 0; r < 8; ++r) {
      const float mn = fmaxf(mrow[r], tmax[r]);
      alpha[r] = __expf(mrow[r] - mn);
      p0[r] = __expf(p0[r] - mn);
      p1[r] = __expf(p1[r] - mn);
      mrow[r] = mn;
    }
    float rs[8];
#pragma unroll
    for (int r = 0; r < 8; ++r) rs[r] = p0[r] + p1[r];
#pragma unroll
    for (int xm = 1; xm < 16; xm <<= 1)
#pragma unroll
      for (int r = 0; r < 8; ++r)
        rs[r] += __shfl_xor(rs[r], xm, 32);
#pragma unroll
    for (int r = 0; r < 8; ++r) lrow[r] = lrow[r] * alpha[r] + rs[r];
#pragma unroll
    for (int t = 0; t < 4; ++t)
#pragma unroll
      for (int r = 0; r < 8; ++r) o[t][r] *= alpha[r];

    // C-layout P -> LDS (row-major 16x32 bf16)
#pragma unroll
    for (int r = 0; r < 8; ++r) {
      const int rowg = hi * 8 + r;
      P[rowg * 32 + c16]      = f32_to_bf16(p0[r]);
      P[rowg * 32 + 16 + c16] = f32_to_bf16(p1[r]);
    }
    // wave-local store->load ordering on the DS counter
    asm volatile("s_wait_dscnt 0" ::: "memory");
    // reload as A-fragment (row = lane&15, K halves by hi)
    v16bf pf = load_frag_bf16(P + c16 * 32 + koff);

    // O += P * V  (V stored transposed: row d, contiguous k)
#pragma unroll
    for (int t = 0; t < 4; ++t) {
      v16bf vf = load_frag_bf16(Vb + (t * 16 + c16) * T_SEQ + kt + koff);
      o[t] = __builtin_amdgcn_wmma_f32_16x16x32_bf16(false, pf, false, vf, (short)0, o[t], false, false);
    }
  }

  // finalize: O /= l, write att[b, t, h*64 + d] as bf16 (token-major for final GEMM)
#pragma unroll
  for (int t = 0; t < 4; ++t) {
#pragma unroll
    for (int r = 0; r < 8; ++r) {
      const int rowg = q0 + hi * 8 + r;
      const float v = o[t][r] / lrow[r];
      att[(size_t)(b * T_SEQ + rowg) * D_MODEL + h * HEAD_DIM + t * 16 + c16] = f32_to_bf16(v);
    }
  }
}

// ---------- host ----------
static inline void launch_cvt(const float* src, u16* dst, int n, hipStream_t s) {
  const int n8 = n / 8;
  cvt_bf16_k<<<dim3((n8 + 255) / 256), dim3(256), 0, s>>>(src, dst, n8);
}

extern "C" void kernel_launch(void* const* d_in, const int* in_sizes, int n_in,
                              void* d_out, int out_size, void* d_ws, size_t ws_size,
                              hipStream_t stream)
{
  (void)in_sizes; (void)n_in; (void)out_size; (void)ws_size;

  const float* query = (const float*)d_in[0];
  const float* key   = (const float*)d_in[1];
  const float* value = (const float*)d_in[2];
  const int*   mask  = (const int*)  d_in[3];
  const float* wq = (const float*)d_in[4];  const float* bq = (const float*)d_in[5];
  const float* wk = (const float*)d_in[6];  const float* bk = (const float*)d_in[7];
  const float* wv = (const float*)d_in[8];  const float* bv = (const float*)d_in[9];
  const float* wo = (const float*)d_in[10]; const float* bo = (const float*)d_in[11];

  char* ws = (char*)d_ws;
  const size_t seg_x = (size_t)M_TOK * D_MODEL * sizeof(u16);    // 8 MiB
  const size_t seg_w = (size_t)D_MODEL * D_MODEL * sizeof(u16);  // 2 MiB
  u16* Xq  = (u16*)(ws + 0 * seg_x);
  u16* Xk  = (u16*)(ws + 1 * seg_x);
  u16* Xv  = (u16*)(ws + 2 * seg_x);
  char* wsw = ws + 3 * seg_x;
  u16* Wqb = (u16*)(wsw + 0 * seg_w);
  u16* Wkb = (u16*)(wsw + 1 * seg_w);
  u16* Wvb = (u16*)(wsw + 2 * seg_w);
  u16* Wob = (u16*)(wsw + 3 * seg_w);
  char* wsh = wsw + 4 * seg_w;
  u16* Qh  = (u16*)(wsh + 0 * seg_x);
  u16* Kh  = (u16*)(wsh + 1 * seg_x);
  u16* Vt  = (u16*)(wsh + 2 * seg_x);
  u16* att = (u16*)(wsh + 3 * seg_x);
  u32* mbw = (u32*)(wsh + 4 * seg_x);    // packed mask bits, 1 MiB

  const int n_act = M_TOK * D_MODEL;     // 4194304
  const int n_w   = D_MODEL * D_MODEL;   // 1048576
  launch_cvt(query, Xq, n_act, stream);
  launch_cvt(key,   Xk, n_act, stream);
  launch_cvt(value, Xv, n_act, stream);
  launch_cvt(wq, Wqb, n_w, stream);
  launch_cvt(wk, Wkb, n_w, stream);
  launch_cvt(wv, Wvb, n_w, stream);
  launch_cvt(wo, Wob, n_w, stream);

  const int nwords = B_SZ * T_SEQ * TW;  // 262144
  mask_pack_k<<<dim3((nwords * 32) / 256), dim3(256), 0, stream>>>(mask, mbw, nwords);

  dim3 gg(D_MODEL / 128, M_TOK / 128);   // (8, 32) blocks of 256 threads
  dim3 bb(256);
  gemm_wmma_k<0><<<gg, bb, 0, stream>>>(Xq, Wqb, bq, Qh);
  gemm_wmma_k<0><<<gg, bb, 0, stream>>>(Xk, Wkb, bk, Kh);
  gemm_wmma_k<1><<<gg, bb, 0, stream>>>(Xv, Wvb, bv, Vt);

  const int n_waves = B_SZ * N_HEADS * (T_SEQ / 16);  // 4096
  flash_attn_k<<<dim3(n_waves / 8), bb, 0, stream>>>(Qh, Kh, Vt, mbw, att);

  gemm_wmma_k<2><<<gg, bb, 0, stream>>>(att, Wob, bo, d_out);
}
